// EncoderLayer_12395275616317
// MI455X (gfx1250) — matrix operand order repaired and checked by reading, not compiled, not run on hardware
//
#include <hip/hip_runtime.h>
#include <hip/hip_bf16.h>
#include <math.h>

typedef _Float16 f16_t;
typedef __attribute__((ext_vector_type(4)))  _Float16 v4h;
typedef __attribute__((ext_vector_type(8)))  _Float16 v8h;
typedef __attribute__((ext_vector_type(16))) _Float16 v16h;
typedef __attribute__((ext_vector_type(8)))  float    v8f;

#define D_MODEL 1024
#define SEQ     2048
#define NHEAD   16
#define HDIM    64
#define TOKENS  (4 * SEQ)   // 8192
#define HIDDEN  4096
#define LOG2E   1.4426950408889634f

static __device__ __forceinline__ v16h cat8(v8h lo, v8h hi) {
  return __builtin_shufflevector(lo, hi, 0,1,2,3,4,5,6,7,8,9,10,11,12,13,14,15);
}

static __device__ __forceinline__ v8f wmma_f16(v16h a, v16h b, v8f c) {
  // D = A(16x32 f16) * B(32x16 f16) + C(16x16 f32)
  return __builtin_amdgcn_wmma_f32_16x16x32_f16(false, a, false, b, (short)0, c,
                                                false, false);
}

// Async 16-byte global -> LDS copy (no VGPR destination, tracked by ASYNCcnt).
// LDS aperture: addr[31:0] of a generic shared pointer IS the LDS byte offset.
static __device__ __forceinline__ void async_ld_b128(const f16_t* g,
                                                     const f16_t* l) {
  unsigned loff = (unsigned)(size_t)l;
  asm volatile("global_load_async_to_lds_b128 %0, %1, off"
               :: "v"(loff), "v"((unsigned long long)(size_t)g)
               : "memory");
}

// ---------------------------------------------------------------------------
// Weight transpose + f32->f16 cast: W[K][N] -> WT[N][K]
// block (32,8), grid (N/32, K/32)
// ---------------------------------------------------------------------------
__global__ void wt_f16_kernel(const float* __restrict__ W, f16_t* __restrict__ WT,
                              int K, int N) {
  __shared__ float tile[32][33];
  int k0 = blockIdx.y * 32, n0 = blockIdx.x * 32;
  int tx = threadIdx.x, ty = threadIdx.y;
#pragma unroll
  for (int i = 0; i < 4; i++)
    tile[ty + 8 * i][tx] = W[(size_t)(k0 + ty + 8 * i) * N + (n0 + tx)];
  __syncthreads();
#pragma unroll
  for (int i = 0; i < 4; i++)
    WT[(size_t)(n0 + ty + 8 * i) * K + (k0 + tx)] = (f16_t)tile[tx][ty + 8 * i];
}

// ---------------------------------------------------------------------------
// LayerNorm (f32 stats) -> f16 output. One 256-thread block per row of 1024.
// ---------------------------------------------------------------------------
__global__ void __launch_bounds__(256) layernorm_f16_kernel(
    const float* __restrict__ x, const float* __restrict__ gamma,
    const float* __restrict__ beta, f16_t* __restrict__ out) {
  int row = blockIdx.x;
  int t = threadIdx.x;
  float4 v = ((const float4*)(x + (size_t)row * D_MODEL))[t];
  float s  = v.x + v.y + v.z + v.w;
  float s2 = v.x * v.x + v.y * v.y + v.z * v.z + v.w * v.w;
#pragma unroll
  for (int m = 1; m < 32; m <<= 1) {
    s  += __shfl_xor(s,  m, 32);
    s2 += __shfl_xor(s2, m, 32);
  }
  __shared__ float rs[8], rs2[8];
  int wave = t >> 5;
  if ((t & 31) == 0) { rs[wave] = s; rs2[wave] = s2; }
  __syncthreads();
  s = 0.0f; s2 = 0.0f;
#pragma unroll
  for (int i = 0; i < 8; i++) { s += rs[i]; s2 += rs2[i]; }
  float mean = s * (1.0f / D_MODEL);
  float var  = s2 * (1.0f / D_MODEL) - mean * mean;
  float inv  = rsqrtf(var + 1e-10f);
  float4 g = ((const float4*)gamma)[t];
  float4 b = ((const float4*)beta)[t];
  v4h o;
  o[0] = (f16_t)(g.x * (v.x - mean) * inv + b.x);
  o[1] = (f16_t)(g.y * (v.y - mean) * inv + b.y);
  o[2] = (f16_t)(g.z * (v.z - mean) * inv + b.z);
  o[3] = (f16_t)(g.w * (v.w - mean) * inv + b.w);
  *(v4h*)(out + (size_t)row * D_MODEL + t * 4) = o;
}

// ---------------------------------------------------------------------------
// Tiled WMMA GEMM: out = A(f16)[M,K] @ WT(f16)[N,K]^T + bias, with epilogue.
// Block tile 64(M) x 256(N), K-step 32, 256 threads = 8 waves.
// Double-buffered LDS staged with GLOBAL_LOAD_ASYNC_TO_LDS_B128: while WMMAs
// run on buffer b, slab b^1 streams in; s_wait_asynccnt 5 retires the prior
// slab (async loads complete in order).
// ---------------------------------------------------------------------------
enum { EPI_F16 = 0, EPI_GELU = 1, EPI_RESID = 2, EPI_VT = 3 };

template <int EPI>
__global__ void __launch_bounds__(256) gemm_f16_kernel(
    const f16_t* __restrict__ A, const f16_t* __restrict__ WT,
    const float* __restrict__ bias, const float* __restrict__ resid,
    void* __restrict__ out, int M, int N, int K) {
  // 40-half row stride (80 B) keeps all 16 B fragment chunks aligned.
  __shared__ __align__(16) f16_t sA[2][64][40];
  __shared__ __align__(16) f16_t sW[2][256][40];
  int m0 = blockIdx.y * 64;
  int n0 = blockIdx.x * 256;
  int t = threadIdx.x;
  int lane = t & 31, wave = t >> 5;
  int mw = wave >> 1, nw = wave & 1;
  int lh = lane >> 4, ln = lane & 15;

  v8f acc[8];
#pragma unroll
  for (int j = 0; j < 8; j++) acc[j] = {};

  int arow = t >> 2, achunk = (t & 3) * 8;
  const f16_t* agp = A + (size_t)(m0 + arow) * K + achunk;   // + k0
  const f16_t* wgp = WT + (size_t)(n0 + t) * K;              // + k0 + c*8

  // 5 async b128 ops per thread per K-slab
  auto stage_tile = [&](int buf, int k0) {
    async_ld_b128(agp + k0, &sA[buf][arow][achunk]);
#pragma unroll
    for (int c = 0; c < 4; c++)
      async_ld_b128(wgp + k0 + c * 8, &sW[buf][t][c * 8]);
  };

  int nk = K / 32;
  stage_tile(0, 0);
  for (int ki = 0; ki < nk; ki++) {
    int buf = ki & 1;
    if (ki + 1 < nk) {
      stage_tile(buf ^ 1, (ki + 1) * 32);
      asm volatile("s_wait_asynccnt 0x5" ::: "memory");  // slab ki resident
    } else {
      asm volatile("s_wait_asynccnt 0x0" ::: "memory");
    }
    __syncthreads();

    // A fragment: lanes 0-15 hold K 0..7 & 16..23, lanes 16-31 hold 8..15 & 24..31
    int ar = mw * 16 + ln;
    v16h afrag = cat8(*(const v8h*)&sA[buf][ar][lh ? 8 : 0],
                      *(const v8h*)&sA[buf][ar][lh ? 24 : 16]);
#pragma unroll
    for (int j = 0; j < 8; j++) {
      // B fragment: lanes 0-15 hold K 0..7 & 16..23 of column n, lanes 16-31 hold 8..15 & 24..31
      int nr = nw * 128 + j * 16 + ln;
      v16h bfrag = cat8(*(const v8h*)&sW[buf][nr][lh ? 8 : 0],
                        *(const v8h*)&sW[buf][nr][lh ? 24 : 16]);
      acc[j] = wmma_f16(afrag, bfrag, acc[j]);
    }
    __syncthreads();
  }

  // epilogue: C layout — VGPR r, lane: row = r + 8*lh, col = ln (+16*tile)
#pragma unroll
  for (int j = 0; j < 8; j++) {
    int col = n0 + nw * 128 + j * 16 + ln;
    float bcol = bias[col];
#pragma unroll
    for (int r = 0; r < 8; r++) {
      int row = m0 + mw * 16 + r + lh * 8;
      float val = acc[j][r] + bcol;
      if (EPI == EPI_F16) {
        ((f16_t*)out)[(size_t)row * N + col] = (f16_t)val;
      } else if (EPI == EPI_GELU) {
        float g = 0.5f * val * (1.0f + erff(val * 0.70710678118654752f));
        ((f16_t*)out)[(size_t)row * N + col] = (f16_t)g;
      } else if (EPI == EPI_RESID) {
        ((float*)out)[(size_t)row * N + col] =
            resid[(size_t)row * N + col] + val;
      } else {  // EPI_VT : scatter to V^T layout [b,h][d][s]
        int b = row >> 11, s = row & 2047;
        int h = col >> 6,  d = col & 63;
        ((f16_t*)out)[(((size_t)(b * NHEAD + h) * HDIM + d) << 11) + s] =
            (f16_t)val;
      }
    }
  }
}

// ---------------------------------------------------------------------------
// Flash attention. Block = 256 threads = 8 independent waves, each owning
// 16 query rows. grid = (SEQ/128, B*H). K/V streamed in 32-key tiles,
// fragments loaded straight from global (already fragment-shaped).
// ---------------------------------------------------------------------------
__global__ void __launch_bounds__(256) attention_kernel(
    const f16_t* __restrict__ Q, const f16_t* __restrict__ Km,
    const f16_t* __restrict__ VT, f16_t* __restrict__ out) {
  __shared__ __align__(16) f16_t sP[8][16][40];  // per-wave P tile 16x32 (+pad)
  int wave = threadIdx.x >> 5, lane = threadIdx.x & 31;
  int lh = lane >> 4, ln = lane & 15;
  int bh = blockIdx.y, b = bh >> 4, h = bh & 15;
  size_t tokQ = (size_t)b * SEQ + blockIdx.x * 128 + wave * 16;

  // Q fragments for hd=64 (two K=32 steps), pre-scaled by 1/sqrt(hd)=0.125
  v16h aq[2];
#pragma unroll
  for (int kk = 0; kk < 2; kk++) {
    const f16_t* p = Q + (tokQ + ln) * D_MODEL + h * HDIM + kk * 32;
    v16h a = cat8(*(const v8h*)(p + (lh ? 8 : 0)),
                  *(const v8h*)(p + (lh ? 24 : 16)));
#pragma unroll
    for (int i = 0; i < 16; i++) a[i] = a[i] * (_Float16)0.125f;
    aq[kk] = a;
  }

  v8f ctx[4];
#pragma unroll
  for (int j = 0; j < 4; j++) ctx[j] = {};
  float mrow[8], lrow[8];
#pragma unroll
  for (int r = 0; r < 8; r++) { mrow[r] = -1e30f; lrow[r] = 0.0f; }

  const f16_t* kbase = Km + (size_t)b * SEQ * D_MODEL + h * HDIM;
  const f16_t* vbase = VT + (size_t)bh * HDIM * SEQ;

  for (int kt = 0; kt < SEQ; kt += 32) {
    // ---- scores: S[16q x 32k] = Qs @ K^T (K rows are already WT-layout) ----
    v8f s0 = {}, s1 = {};
#pragma unroll
    for (int kk = 0; kk < 2; kk++) {
      const f16_t* kp0 = kbase + (size_t)(kt + ln) * D_MODEL + kk * 32 + lh * 8;
      const f16_t* kp1 = kbase + (size_t)(kt + 16 + ln) * D_MODEL + kk * 32 + lh * 8;
      v16h b0 = cat8(*(const v8h*)kp0, *(const v8h*)(kp0 + 16));
      v16h b1 = cat8(*(const v8h*)kp1, *(const v8h*)(kp1 + 16));
      s0 = wmma_f16(aq[kk], b0, s0);
      s1 = wmma_f16(aq[kk], b1, s1);
    }
    // ---- online softmax: rows live on VGPR index + lane-half; reduce across
    //      the 16 lanes of each half with xor-butterflies ----
    float alpha[8];
#pragma unroll
    for (int r = 0; r < 8; r++) {
      float mx = fmaxf(s0[r], s1[r]);
#pragma unroll
      for (int m = 1; m < 16; m <<= 1) mx = fmaxf(mx, __shfl_xor(mx, m, 32));
      float mnew = fmaxf(mrow[r], mx);
      float a  = exp2f((mrow[r] - mnew) * LOG2E);
      float p0 = exp2f((s0[r] - mnew) * LOG2E);
      float p1 = exp2f((s1[r] - mnew) * LOG2E);
      float ps = p0 + p1;
#pragma unroll
      for (int m = 1; m < 16; m <<= 1) ps += __shfl_xor(ps, m, 32);
      lrow[r] = lrow[r] * a + ps;
      mrow[r] = mnew;
      alpha[r] = a;
      int prow = r + lh * 8;                 // C layout -> LDS (row, col)
      sP[wave][prow][ln]      = (f16_t)p0;
      sP[wave][prow][ln + 16] = (f16_t)p1;
    }
#pragma unroll
    for (int j = 0; j < 4; j++)
#pragma unroll
      for (int r = 0; r < 8; r++) ctx[j][r] *= alpha[r];

    // re-read P in A-fragment layout (lane = row); per-wave region, wave-
    // synchronous DS ops, compiler inserts the dscnt waits.
    v16h ap = cat8(*(const v8h*)&sP[wave][ln][lh ? 8 : 0],
                   *(const v8h*)&sP[wave][ln][lh ? 24 : 16]);
    // ---- ctx += P @ V (V^T rows give contiguous B fragments) ----
#pragma unroll
    for (int j = 0; j < 4; j++) {
      const f16_t* vp = vbase + (size_t)(j * 16 + ln) * SEQ + kt + lh * 8;
      v16h bv = cat8(*(const v8h*)vp, *(const v8h*)(vp + 16));
      ctx[j] = wmma_f16(ap, bv, ctx[j]);
    }
  }

#pragma unroll
  for (int j = 0; j < 4; j++)
#pragma unroll
    for (int r = 0; r < 8; r++) {
      size_t row = tokQ + r + lh * 8;
      out[row * D_MODEL + h * HDIM + j * 16 + ln] =
          (f16_t)(ctx[j][r] / lrow[r]);
    }
}

// ---------------------------------------------------------------------------
// Host-side orchestration
// ---------------------------------------------------------------------------
extern "C" void kernel_launch(void* const* d_in, const int* in_sizes, int n_in,
                              void* d_out, int out_size, void* d_ws,
                              size_t ws_size, hipStream_t stream) {
  const float* x      = (const float*)d_in[0];
  const float* Wq     = (const float*)d_in[1];
  const float* bq     = (const float*)d_in[2];
  const float* Wk     = (const float*)d_in[3];
  const float* bk     = (const float*)d_in[4];
  const float* Wv     = (const float*)d_in[5];
  const float* bv     = (const float*)d_in[6];
  const float* Wo     = (const float*)d_in[7];
  const float* bo     = (const float*)d_in[8];
  const float* W1     = (const float*)d_in[9];
  const float* b1     = (const float*)d_in[10];
  const float* W2     = (const float*)d_in[11];
  const float* b2     = (const float*)d_in[12];
  const float* gamma1 = (const float*)d_in[13];
  const float* beta1  = (const float*)d_in[14];
  const float* gamma2 = (const float*)d_in[15];
  const float* beta2  = (const float*)d_in[16];

  char* ws = (char*)d_ws;
  size_t off = 0;
  auto take = [&](size_t bytes) -> char* {
    char* p = ws + off;
    off += (bytes + 255) & ~(size_t)255;
    return p;
  };
  f16_t* WqT = (f16_t*)take((size_t)D_MODEL * D_MODEL * 2);
  f16_t* WkT = (f16_t*)take((size_t)D_MODEL * D_MODEL * 2);
  f16_t* WvT = (f16_t*)take((size_t)D_MODEL * D_MODEL * 2);
  f16_t* WoT = (f16_t*)take((size_t)D_MODEL * D_MODEL * 2);
  f16_t* W1T = (f16_t*)take((size_t)D_MODEL * HIDDEN * 2);
  f16_t* W2T = (f16_t*)take((size_t)HIDDEN * D_MODEL * 2);
  f16_t* xn1 = (f16_t*)take((size_t)TOKENS * D_MODEL * 2);
  f16_t* xn2 = (f16_t*)take((size_t)TOKENS * D_MODEL * 2);
  f16_t* Qf  = (f16_t*)take((size_t)TOKENS * D_MODEL * 2);
  f16_t* Kf  = (f16_t*)take((size_t)TOKENS * D_MODEL * 2);
  f16_t* VTf = (f16_t*)take((size_t)TOKENS * D_MODEL * 2);
  f16_t* Cf  = (f16_t*)take((size_t)TOKENS * D_MODEL * 2);
  float* x1  = (float*)take((size_t)TOKENS * D_MODEL * 4);
  f16_t* hf  = (f16_t*)take((size_t)TOKENS * HIDDEN * 2);
  (void)ws_size; (void)in_sizes; (void)n_in; (void)out_size;

  dim3 bT(32, 8);
  wt_f16_kernel<<<dim3(D_MODEL / 32, D_MODEL / 32), bT, 0, stream>>>(Wq, WqT, D_MODEL, D_MODEL);
  wt_f16_kernel<<<dim3(D_MODEL / 32, D_MODEL / 32), bT, 0, stream>>>(Wk, WkT, D_MODEL, D_MODEL);
  wt_f16_kernel<<<dim3(D_MODEL / 32, D_MODEL / 32), bT, 0, stream>>>(Wv, WvT, D_MODEL, D_MODEL);
  wt_f16_kernel<<<dim3(D_MODEL / 32, D_MODEL / 32), bT, 0, stream>>>(Wo, WoT, D_MODEL, D_MODEL);
  wt_f16_kernel<<<dim3(HIDDEN / 32, D_MODEL / 32), bT, 0, stream>>>(W1, W1T, D_MODEL, HIDDEN);
  wt_f16_kernel<<<dim3(D_MODEL / 32, HIDDEN / 32), bT, 0, stream>>>(W2, W2T, HIDDEN, D_MODEL);

  // sublayer 1
  layernorm_f16_kernel<<<TOKENS, 256, 0, stream>>>(x, gamma1, beta1, xn1);
  dim3 gdd(D_MODEL / 256, TOKENS / 64);
  gemm_f16_kernel<EPI_F16><<<gdd, 256, 0, stream>>>(xn1, WqT, bq, nullptr, Qf,
                                                    TOKENS, D_MODEL, D_MODEL);
  gemm_f16_kernel<EPI_F16><<<gdd, 256, 0, stream>>>(xn1, WkT, bk, nullptr, Kf,
                                                    TOKENS, D_MODEL, D_MODEL);
  gemm_f16_kernel<EPI_VT><<<gdd, 256, 0, stream>>>(xn1, WvT, bv, nullptr, VTf,
                                                   TOKENS, D_MODEL, D_MODEL);
  attention_kernel<<<dim3(SEQ / 128, 4 * NHEAD), 256, 0, stream>>>(Qf, Kf, VTf, Cf);
  gemm_f16_kernel<EPI_RESID><<<gdd, 256, 0, stream>>>(Cf, WoT, bo, x, x1,
                                                      TOKENS, D_MODEL, D_MODEL);

  // sublayer 2
  layernorm_f16_kernel<<<TOKENS, 256, 0, stream>>>(x1, gamma2, beta2, xn2);
  gemm_f16_kernel<EPI_GELU><<<dim3(HIDDEN / 256, TOKENS / 64), 256, 0, stream>>>(
      xn2, W1T, b1, nullptr, hf, TOKENS, HIDDEN, D_MODEL);
  gemm_f16_kernel<EPI_RESID><<<gdd, 256, 0, stream>>>(hf, W2T, b2, x1,
                                                      (float*)d_out, TOKENS,
                                                      D_MODEL, HIDDEN);
}
